// HSTULayer_36928128811110
// MI455X (gfx1250) — matrix operand order repaired and checked by reading, not compiled, not run on hardware
//
#include <hip/hip_runtime.h>
#include <math.h>

// ---------------------------------------------------------------------------
// Transformer layer for MI455X (gfx1250):
//  - all matmuls via v_wmma_f32_16x16x32_bf16 (128x64 block tiles, 8 waves)
//  - weight GEMMs use GLOBAL_LOAD_ASYNC_TO_LDS_B128 + s_wait_asynccnt with
//    double-buffered LDS; attention kernels convert f32->bf16 in a
//    register-prefetch pipeline.
// ---------------------------------------------------------------------------

typedef __attribute__((ext_vector_type(16))) __bf16 v16bf;
typedef __attribute__((ext_vector_type(8)))  float  v8f;

#define S_DIM 256
#define B_DIM 32
#define H_DIM 512
#define P_DIM 2048
#define T_DIM (P_DIM + S_DIM)   // 2304
#define FF_DIM (4 * H_DIM)      // 2048
#define M_ROWS (S_DIM * B_DIM)  // 8192

#define USE_ASYNC_COPY 1

__device__ __forceinline__ unsigned short f2bf(float f) {
    unsigned int x = __float_as_uint(f);
    unsigned int r = x + 0x7FFFu + ((x >> 16) & 1u);   // round-to-nearest-even
    return (unsigned short)(r >> 16);
}
__device__ __forceinline__ unsigned int pack2(float a, float b) {
    return (unsigned int)f2bf(a) | ((unsigned int)f2bf(b) << 16);
}

union FragU { v16bf v; unsigned int u[8]; };

// A fragment (16x32 bf16, MxK): lanes 0-15 hold K0-15, lanes 16-31 K16-31
__device__ __forceinline__ v16bf frag_ld_a(const unsigned short (*tile)[32], int rowbase, int lane) {
    FragU f;
    const unsigned int* r = reinterpret_cast<const unsigned int*>(tile[rowbase + (lane & 15)]);
    int kb = (lane >> 4) << 3;            // 0 or 8 dwords
#pragma unroll
    for (int i = 0; i < 8; ++i) f.u[i] = r[kb + i];
    return f.v;
}
// B fragment (32x16 bf16, KxN) stored N-major (tile[n][k]): lanes 0-15 K0-15, lanes 16-31 K16-31
__device__ __forceinline__ v16bf frag_ld_b(const unsigned short (*tile)[32], int rowbase, int lane) {
    FragU f;
    const unsigned int* r = reinterpret_cast<const unsigned int*>(tile[rowbase + (lane & 15)]);
    int kb = (lane >> 4) << 3;            // 0 or 8 dwords
#pragma unroll
    for (int i = 0; i < 8; ++i) f.u[i] = r[kb + i];
    return f.v;
}
__device__ __forceinline__ v8f wmma_bf16(v16bf a, v16bf b, v8f c) {
    return __builtin_amdgcn_wmma_f32_16x16x32_bf16(false, a, false, b, (short)0, c, false, false);
}

#if USE_ASYNC_COPY
__device__ __forceinline__ void async_b128(unsigned lds_off, const void* g) {
    asm volatile("global_load_async_to_lds_b128 %0, %1, off" :: "v"(lds_off), "v"(g) : "memory");
}
__device__ __forceinline__ void wait_async3() { asm volatile("s_wait_asynccnt 0x3" ::: "memory"); }
__device__ __forceinline__ void wait_async0() { asm volatile("s_wait_asynccnt 0x0" ::: "memory"); }
#endif

// ---------------------------------------------------------------------------
// Layout converters
// ---------------------------------------------------------------------------
__global__ __launch_bounds__(256) void cvt_hidden_kernel(const float* __restrict__ hidden, // [S,B,H]
                                                         float* __restrict__ hF,           // [B*S,H]
                                                         unsigned short* __restrict__ hB) {
    int idx = blockIdx.x * 256 + threadIdx.x;
    int i = idx * 4;
    int h = i % H_DIM;
    int sb = i / H_DIM;
    int b = sb % B_DIM;
    int s = sb / B_DIM;
    float4 v = *reinterpret_cast<const float4*>(hidden + i);
    size_t o = ((size_t)b * S_DIM + s) * H_DIM + h;
    *reinterpret_cast<float4*>(hF + o) = v;
    uint2 p; p.x = pack2(v.x, v.y); p.y = pack2(v.z, v.w);
    *reinterpret_cast<uint2*>(hB + o) = p;
}

// W [K,N] f32 -> Wt [N,K] bf16
__global__ __launch_bounds__(256) void cvt_wt_kernel(const float* __restrict__ W,
                                                     unsigned short* __restrict__ Wt,
                                                     int K, int N) {
    int idx = blockIdx.x * 256 + threadIdx.x;
    int k = idx % K, n = idx / K;
    Wt[idx] = f2bf(W[(size_t)k * N + n]);
}

// ---------------------------------------------------------------------------
// Generic bf16 WMMA GEMM, 128x64 tile: C = A[MxK] * Wt[NxK]^T + bias (+epilogues)
// ---------------------------------------------------------------------------
__global__ __launch_bounds__(256) void gemm_kernel(
    const unsigned short* __restrict__ A,    // [M,K] bf16
    const unsigned short* __restrict__ Wt,   // [N,K] bf16 (pre-transposed weights)
    const float* __restrict__ bias,          // [N]
    const float* __restrict__ resid,         // [M,N] f32 or nullptr
    float* __restrict__ outF,                // [M,N] f32 or nullptr
    unsigned short* __restrict__ outB,       // [M,N] bf16 or nullptr
    float* __restrict__ outSBH,              // [S,B,H] f32 or nullptr (requires N==H)
    int M, int N, int K, int doGelu) {
#if USE_ASYNC_COPY
    __shared__ __align__(16) unsigned short As[2][128][32];
    __shared__ __align__(16) unsigned short Bs[2][64][32];
#else
    __shared__ __align__(16) unsigned short As[1][128][32];
    __shared__ __align__(16) unsigned short Bs[1][64][32];
#endif
    const int tid = threadIdx.x, lane = tid & 31, wave = tid >> 5;
    const int m0 = blockIdx.y * 128, n0 = blockIdx.x * 64;
    const int arow = tid >> 1, ahalf = tid & 1;      // A staging: 32B per thread
    const int brow = tid >> 2, bseg = tid & 3;       // B staging: 16B per thread
    const unsigned short* Ag = A + (size_t)(m0 + arow) * K + ahalf * 16;
    const unsigned short* Bg = Wt + (size_t)(n0 + brow) * K + bseg * 8;
    v8f acc[4] = {};
    const int steps = K >> 5;

#if USE_ASYNC_COPY
    const unsigned lA0 = (unsigned)(size_t)&As[0][arow][ahalf * 16];
    const unsigned lA1 = (unsigned)(size_t)&As[1][arow][ahalf * 16];
    const unsigned lB0 = (unsigned)(size_t)&Bs[0][brow][bseg * 8];
    const unsigned lB1 = (unsigned)(size_t)&Bs[1][brow][bseg * 8];
    // prologue: fill buffer 0
    async_b128(lA0, Ag);
    async_b128(lA0 + 16, Ag + 8);
    async_b128(lB0, Bg);
    for (int it = 0; it < steps; ++it) {
        const int cur = it & 1;
        if (it + 1 < steps) {       // fill the other buffer while computing
            int k0 = (it + 1) << 5;
            unsigned la = cur ? lA0 : lA1;
            unsigned lb = cur ? lB0 : lB1;
            async_b128(la, Ag + k0);
            async_b128(la + 16, Ag + k0 + 8);
            async_b128(lb, Bg + k0);
            wait_async3();          // current buffer's 3 asyncs retired (in-order)
        } else {
            wait_async0();
        }
        __syncthreads();
        v16bf af = frag_ld_a(As[cur], wave << 4, lane);
#pragma unroll
        for (int ni = 0; ni < 4; ++ni)
            acc[ni] = wmma_bf16(af, frag_ld_b(Bs[cur], ni << 4, lane), acc[ni]);
        __syncthreads();
    }
#else
    uint4 a0 = *reinterpret_cast<const uint4*>(Ag);
    uint4 a1 = *reinterpret_cast<const uint4*>(Ag + 8);
    uint4 b0 = *reinterpret_cast<const uint4*>(Bg);
    for (int it = 0; it < steps; ++it) {
        *reinterpret_cast<uint4*>(&As[0][arow][ahalf * 16]) = a0;
        *reinterpret_cast<uint4*>(&As[0][arow][ahalf * 16 + 8]) = a1;
        *reinterpret_cast<uint4*>(&Bs[0][brow][bseg * 8]) = b0;
        __syncthreads();
        if (it + 1 < steps) {       // register prefetch of next tile
            int k0 = (it + 1) << 5;
            a0 = *reinterpret_cast<const uint4*>(Ag + k0);
            a1 = *reinterpret_cast<const uint4*>(Ag + k0 + 8);
            b0 = *reinterpret_cast<const uint4*>(Bg + k0);
        }
        v16bf af = frag_ld_a(As[0], wave << 4, lane);
#pragma unroll
        for (int ni = 0; ni < 4; ++ni)
            acc[ni] = wmma_bf16(af, frag_ld_b(Bs[0], ni << 4, lane), acc[ni]);
        __syncthreads();
    }
#endif

    const int rbase = m0 + (wave << 4) + ((lane >> 4) << 3);
#pragma unroll
    for (int ni = 0; ni < 4; ++ni) {
        const int c = n0 + (ni << 4) + (lane & 15);
        const float bi = bias[c];
#pragma unroll
        for (int r = 0; r < 8; ++r) {
            int m = rbase + r;
            float v = acc[ni][r] + bi;
            if (doGelu) v = 0.5f * v * (1.0f + erff(v * 0.70710678118654752f));
            if (resid) v += resid[(size_t)m * N + c];
            if (outF)  outF[(size_t)m * N + c] = v;
            if (outB)  outB[(size_t)m * N + c] = f2bf(v);
            if (outSBH) {
                int b = m / S_DIM, s = m % S_DIM;
                outSBH[((size_t)s * B_DIM + b) * H_DIM + c] = v;
            }
        }
    }
}

// ---------------------------------------------------------------------------
// scores[b,s,t] = (q[b,s,:] . k_all[b,t,:]) / sqrt(H), masked. 128x64 tiles.
// ---------------------------------------------------------------------------
__global__ __launch_bounds__(256) void scores_kernel(
    const unsigned short* __restrict__ Qb,   // [B*S,H] bf16
    const float* __restrict__ pastK,         // [B,P,H] f32
    const float* __restrict__ newKf,         // [B*S,H] f32
    const int* __restrict__ pastLens,        // [B]
    float* __restrict__ scores) {            // [B,S,T] f32
    __shared__ __align__(16) unsigned short As[128][32];
    __shared__ __align__(16) unsigned short Bs[64][32];
    const int tid = threadIdx.x, lane = tid & 31, wave = tid >> 5;
    const int b = blockIdx.z;
    const int s0 = blockIdx.y * 128, t0 = blockIdx.x * 64;
    const int arow = tid >> 1, ahalf = tid & 1;
    const int brow = tid >> 2, bseg = tid & 3;
    v8f acc[4] = {};

    const unsigned short* Ag = Qb + ((size_t)b * S_DIM + s0 + arow) * H_DIM + ahalf * 16;
    const int tRow = t0 + brow;
    const float* krow = (tRow < P_DIM)
        ? (pastK + ((size_t)b * P_DIM + tRow) * H_DIM)
        : (newKf + ((size_t)b * S_DIM + (tRow - P_DIM)) * H_DIM);

    uint4 a0 = *reinterpret_cast<const uint4*>(Ag);
    uint4 a1 = *reinterpret_cast<const uint4*>(Ag + 8);
    float4 f0 = reinterpret_cast<const float4*>(krow)[bseg * 2];
    float4 f1 = reinterpret_cast<const float4*>(krow)[bseg * 2 + 1];

    for (int k0 = 0; k0 < H_DIM; k0 += 32) {
        *reinterpret_cast<uint4*>(&As[arow][ahalf * 16]) = a0;
        *reinterpret_cast<uint4*>(&As[arow][ahalf * 16 + 8]) = a1;
        uint4 pk;
        pk.x = pack2(f0.x, f0.y); pk.y = pack2(f0.z, f0.w);
        pk.z = pack2(f1.x, f1.y); pk.w = pack2(f1.z, f1.w);
        *reinterpret_cast<uint4*>(&Bs[brow][bseg * 8]) = pk;
        __syncthreads();
        if (k0 + 32 < H_DIM) {      // register prefetch of next tile
            a0 = *reinterpret_cast<const uint4*>(Ag + k0 + 32);
            a1 = *reinterpret_cast<const uint4*>(Ag + k0 + 40);
            f0 = reinterpret_cast<const float4*>(krow + k0 + 32)[bseg * 2];
            f1 = reinterpret_cast<const float4*>(krow + k0 + 32)[bseg * 2 + 1];
        }
        v16bf af = frag_ld_a(As, wave << 4, lane);
#pragma unroll
        for (int ni = 0; ni < 4; ++ni)
            acc[ni] = wmma_bf16(af, frag_ld_b(Bs, ni << 4, lane), acc[ni]);
        __syncthreads();
    }

    const float scale = 0.04419417382415922f; // 1/sqrt(512)
    const int plen = pastLens[b];
    const int srow = s0 + (wave << 4) + ((lane >> 4) << 3);
#pragma unroll
    for (int ni = 0; ni < 4; ++ni) {
        const int t = t0 + (ni << 4) + (lane & 15);
#pragma unroll
        for (int r = 0; r < 8; ++r) {
            int s = srow + r;
            bool ok = (t < P_DIM) ? (t < plen) : ((t - P_DIM) <= s);
            scores[((size_t)b * S_DIM + s) * T_DIM + t] = ok ? acc[ni][r] * scale : -INFINITY;
        }
    }
}

// ---------------------------------------------------------------------------
// Row softmax over T, output bf16 probabilities. One wave per row.
// ---------------------------------------------------------------------------
__global__ __launch_bounds__(256) void softmax_kernel(const float* __restrict__ scores,
                                                      unsigned short* __restrict__ probsB) {
    const int lane = threadIdx.x & 31, wave = threadIdx.x >> 5;
    const int row = blockIdx.x * 8 + wave;
    const float* src = scores + (size_t)row * T_DIM;
    float vals[T_DIM / 32];
    float mx = -INFINITY;
#pragma unroll
    for (int i = 0; i < T_DIM / 32; ++i) { vals[i] = src[lane + i * 32]; mx = fmaxf(mx, vals[i]); }
#pragma unroll
    for (int o = 16; o > 0; o >>= 1) mx = fmaxf(mx, __shfl_xor(mx, o, 32));
    float sum = 0.f;
#pragma unroll
    for (int i = 0; i < T_DIM / 32; ++i) { float e = __expf(vals[i] - mx); vals[i] = e; sum += e; }
#pragma unroll
    for (int o = 16; o > 0; o >>= 1) sum += __shfl_xor(sum, o, 32);
    float inv = 1.0f / sum;
    unsigned short* dst = probsB + (size_t)row * T_DIM;
#pragma unroll
    for (int i = 0; i < T_DIM / 32; ++i) dst[lane + i * 32] = f2bf(vals[i] * inv);
}

// ---------------------------------------------------------------------------
// ctx[b,s,h] = sum_t probs[b,s,t] * v_all[b,t,h]. 128x64 tiles, V transposed
// during staging via lane-coalesced strided reads.
// ---------------------------------------------------------------------------
__global__ __launch_bounds__(256) void pv_kernel(
    const unsigned short* __restrict__ probsB,  // [B*S,T] bf16
    const float* __restrict__ pastV,            // [B,P,H]
    const float* __restrict__ newVf,            // [B*S,H]
    unsigned short* __restrict__ ctxB) {        // [B*S,H] bf16
    __shared__ __align__(16) unsigned short As[128][32];
    __shared__ __align__(16) unsigned short Bs[64][32];
    const int tid = threadIdx.x, lane = tid & 31, wave = tid >> 5;
    const int b = blockIdx.z;
    const int s0 = blockIdx.y * 128, h0 = blockIdx.x * 64;
    const int arow = tid >> 1, ahalf = tid & 1;
    const int hcol = tid & 63, kgrp = tid >> 6;    // B staging: 8 strided t's, same h
    v8f acc[4] = {};

    const unsigned short* Ag = probsB + ((size_t)b * S_DIM + s0 + arow) * T_DIM + ahalf * 16;

    auto vload = [&](int k0, float* vv) {
#pragma unroll
        for (int j = 0; j < 8; ++j) {
            int t = k0 + kgrp * 8 + j;
            const float* vr = (t < P_DIM)
                ? (pastV + ((size_t)b * P_DIM + t) * H_DIM)
                : (newVf + ((size_t)b * S_DIM + (t - P_DIM)) * H_DIM);
            vv[j] = vr[h0 + hcol];
        }
    };

    uint4 a0 = *reinterpret_cast<const uint4*>(Ag);
    uint4 a1 = *reinterpret_cast<const uint4*>(Ag + 8);
    float vv[8];
    vload(0, vv);

    for (int k0 = 0; k0 < T_DIM; k0 += 32) {
        *reinterpret_cast<uint4*>(&As[arow][ahalf * 16]) = a0;
        *reinterpret_cast<uint4*>(&As[arow][ahalf * 16 + 8]) = a1;
        uint4 pk;
        pk.x = pack2(vv[0], vv[1]); pk.y = pack2(vv[2], vv[3]);
        pk.z = pack2(vv[4], vv[5]); pk.w = pack2(vv[6], vv[7]);
        *reinterpret_cast<uint4*>(&Bs[hcol][kgrp * 8]) = pk;
        __syncthreads();
        if (k0 + 32 < T_DIM) {      // register prefetch of next tile
            a0 = *reinterpret_cast<const uint4*>(Ag + k0 + 32);
            a1 = *reinterpret_cast<const uint4*>(Ag + k0 + 40);
            vload(k0 + 32, vv);
        }
        v16bf af = frag_ld_a(As, wave << 4, lane);
#pragma unroll
        for (int ni = 0; ni < 4; ++ni)
            acc[ni] = wmma_bf16(af, frag_ld_b(Bs, ni << 4, lane), acc[ni]);
        __syncthreads();
    }

    const int rbase = s0 + (wave << 4) + ((lane >> 4) << 3);
#pragma unroll
    for (int ni = 0; ni < 4; ++ni) {
        const int c = h0 + (ni << 4) + (lane & 15);
#pragma unroll
        for (int r = 0; r < 8; ++r) {
            size_t m = (size_t)b * S_DIM + rbase + r;
            ctxB[m * H_DIM + c] = f2bf(acc[ni][r]);
        }
    }
}

// ---------------------------------------------------------------------------
// Host-side orchestration
// ---------------------------------------------------------------------------
extern "C" void kernel_launch(void* const* d_in, const int* in_sizes, int n_in,
                              void* d_out, int out_size, void* d_ws, size_t ws_size,
                              hipStream_t stream) {
    const float* hidden = (const float*)d_in[0];
    const float* pastK  = (const float*)d_in[1];
    const float* pastV  = (const float*)d_in[2];
    const int*   plens  = (const int*)d_in[3];
    const float* Wq = (const float*)d_in[4],  *bq = (const float*)d_in[5];
    const float* Wk = (const float*)d_in[6],  *bk = (const float*)d_in[7];
    const float* Wv = (const float*)d_in[8],  *bv = (const float*)d_in[9];
    const float* Wo = (const float*)d_in[10], *bo = (const float*)d_in[11];
    const float* W1 = (const float*)d_in[12], *b1 = (const float*)d_in[13];
    const float* W2 = (const float*)d_in[14], *b2 = (const float*)d_in[15];
    float* out = (float*)d_out;

    char* p = (char*)d_ws;
    auto carve = [&](size_t bytes) -> char* {
        char* r = p;
        p += (bytes + 255) & ~(size_t)255;
        return r;
    };
    float*          hF     = (float*)carve((size_t)M_ROWS * H_DIM * 4);
    unsigned short* hB     = (unsigned short*)carve((size_t)M_ROWS * H_DIM * 2);
    unsigned short* qB     = (unsigned short*)carve((size_t)M_ROWS * H_DIM * 2);
    float*          kNewF  = (float*)carve((size_t)M_ROWS * H_DIM * 4);
    float*          vNewF  = (float*)carve((size_t)M_ROWS * H_DIM * 4);
    unsigned short* WqT    = (unsigned short*)carve((size_t)H_DIM * H_DIM * 2);
    unsigned short* WkT    = (unsigned short*)carve((size_t)H_DIM * H_DIM * 2);
    unsigned short* WvT    = (unsigned short*)carve((size_t)H_DIM * H_DIM * 2);
    unsigned short* WoT    = (unsigned short*)carve((size_t)H_DIM * H_DIM * 2);
    unsigned short* W1T    = (unsigned short*)carve((size_t)H_DIM * FF_DIM * 2);
    unsigned short* W2T    = (unsigned short*)carve((size_t)FF_DIM * H_DIM * 2);
    float*          scores = (float*)carve((size_t)M_ROWS * T_DIM * 4);
    unsigned short* probsB = (unsigned short*)carve((size_t)M_ROWS * T_DIM * 2);
    unsigned short* ctxB   = (unsigned short*)carve((size_t)M_ROWS * H_DIM * 2);
    float*          h1F    = (float*)carve((size_t)M_ROWS * H_DIM * 4);
    unsigned short* h1B    = (unsigned short*)carve((size_t)M_ROWS * H_DIM * 2);
    unsigned short* gB     = (unsigned short*)carve((size_t)M_ROWS * FF_DIM * 2);

    // layout conversions
    cvt_hidden_kernel<<<(M_ROWS * H_DIM / 4) / 256, 256, 0, stream>>>(hidden, hF, hB);
    cvt_wt_kernel<<<(H_DIM * H_DIM) / 256, 256, 0, stream>>>(Wq, WqT, H_DIM, H_DIM);
    cvt_wt_kernel<<<(H_DIM * H_DIM) / 256, 256, 0, stream>>>(Wk, WkT, H_DIM, H_DIM);
    cvt_wt_kernel<<<(H_DIM * H_DIM) / 256, 256, 0, stream>>>(Wv, WvT, H_DIM, H_DIM);
    cvt_wt_kernel<<<(H_DIM * H_DIM) / 256, 256, 0, stream>>>(Wo, WoT, H_DIM, H_DIM);
    cvt_wt_kernel<<<(H_DIM * FF_DIM) / 256, 256, 0, stream>>>(W1, W1T, H_DIM, FF_DIM);
    cvt_wt_kernel<<<(FF_DIM * H_DIM) / 256, 256, 0, stream>>>(W2, W2T, FF_DIM, H_DIM);

    dim3 gH(H_DIM / 64, M_ROWS / 128);      // (8,64)
    // q / k_new / v_new projections
    gemm_kernel<<<gH, 256, 0, stream>>>(hB, WqT, bq, nullptr, nullptr, qB, nullptr, M_ROWS, H_DIM, H_DIM, 0);
    gemm_kernel<<<gH, 256, 0, stream>>>(hB, WkT, bk, nullptr, kNewF, nullptr, nullptr, M_ROWS, H_DIM, H_DIM, 0);
    gemm_kernel<<<gH, 256, 0, stream>>>(hB, WvT, bv, nullptr, vNewF, nullptr, nullptr, M_ROWS, H_DIM, H_DIM, 0);

    // attention
    dim3 gS(T_DIM / 64, S_DIM / 128, B_DIM); // (36,2,32)
    scores_kernel<<<gS, 256, 0, stream>>>(qB, pastK, kNewF, plens, scores);
    softmax_kernel<<<M_ROWS / 8, 256, 0, stream>>>(scores, probsB);
    dim3 gP(H_DIM / 64, S_DIM / 128, B_DIM); // (8,2,32)
    pv_kernel<<<gP, 256, 0, stream>>>(probsB, pastV, vNewF, ctxB);

    // O-projection + residual -> h1
    gemm_kernel<<<gH, 256, 0, stream>>>(ctxB, WoT, bo, hF, h1F, h1B, nullptr, M_ROWS, H_DIM, H_DIM, 0);
    // FFN
    dim3 gF(FF_DIM / 64, M_ROWS / 128);     // (32,64)
    gemm_kernel<<<gF, 256, 0, stream>>>(h1B, W1T, b1, nullptr, nullptr, gB, nullptr, M_ROWS, FF_DIM, H_DIM, 1);
    gemm_kernel<<<gH, 256, 0, stream>>>(gB, W2T, b2, h1F, nullptr, nullptr, out, M_ROWS, H_DIM, FF_DIM, 0);
}
